// MSDeformAttn_56118042689544
// MI455X (gfx1250) — matrix operand order repaired and checked by reading, not compiled, not run on hardware
//
#include <hip/hip_runtime.h>
#include <hip/hip_bf16.h>
#include <math.h>

typedef __attribute__((ext_vector_type(16))) _Float16 v16h;
typedef __attribute__((ext_vector_type(8)))  _Float16 v8h;
typedef __attribute__((ext_vector_type(8)))  float    v8f;
typedef __attribute__((ext_vector_type(4)))  float    v4f;

#define NB   4
#define LQn  3840
#define LIN  3840
#define DD   256
#define HH   8
#define LL   4
#define PP   4
#define DHH  32
#define MROWS (NB * LQn)   // 15360

// ---------------------------------------------------------------------------
// LDS-staged GEMM:  C[M x NCOLS] = A[M x 256] * W[256 x NCOLS] + bias
// Block = 256 threads = 8 waves computing a 64x32 macro-tile (4x2 wave grid
// of 16x16 WMMA tiles). K=256 baked in -> fully unrolled, 8 v_wmma per wave.
// Global traffic is 128-bit coalesced; fragments come from LDS (f16).
// Requires M % 64 == 0, NCOLS % 32 == 0 (true at all call sites).
// ---------------------------------------------------------------------------
template <int NCOLS>
__global__ __launch_bounds__(256)
void gemm_wmma_lds(const float* __restrict__ A, const float* __restrict__ W,
                   const float* __restrict__ bias, float* __restrict__ C) {
  constexpr int K = 256;
  __shared__ __align__(16) _Float16 As[64][32];   // [row][k]      4 KB
  __shared__ __align__(16) _Float16 Bs[32][32];   // [col][k] (T)  2 KB

  const int t    = threadIdx.x;
  const int lane = t & 31;
  const int wv   = t >> 5;

  constexpr int NTB = NCOLS / 32;
  const int mt = blockIdx.x / NTB;
  const int nb = blockIdx.x - mt * NTB;
  const int rowBase = mt * 64;
  const int colBase = nb * 32;

  const int hiK  = lane >> 4;       // lane-group selector (ISA 7.12.2)
  const int l16  = lane & 15;
  const int wrow = (wv >> 1) * 16;  // wave's M sub-tile inside macro-tile
  const int wcol = (wv & 1) * 16;   // wave's N sub-tile

  // staging assignments (256 threads)
  const int arow = t >> 2;          // 0..63
  const int akc  = (t & 3) * 8;     // 0,8,16,24
  const int bk   = t >> 3;          // 0..31
  const int bc   = (t & 7) * 4;     // 0..28

  v8f acc = {};

#pragma unroll
  for (int k0 = 0; k0 < K; k0 += 32) {
    // ---- stage A 64x32 chunk: 2 x float4 per thread, coalesced ----
    const float* ag = A + (size_t)(rowBase + arow) * K + k0 + akc;
    v4f a0 = *(const v4f*)ag;
    v4f a1 = *(const v4f*)(ag + 4);
    if (k0 + 32 < K) __builtin_prefetch(ag + 32, 0, 1);   // global_prefetch_b8
    v8h ah;
#pragma unroll
    for (int i = 0; i < 4; ++i) { ah[i] = (_Float16)a0[i]; ah[i + 4] = (_Float16)a1[i]; }
    *(v8h*)&As[arow][akc] = ah;                           // 16-B LDS store

    // ---- stage B 32x32 chunk transposed: 1 x float4 per thread ----
    const float* bg = W + (size_t)(k0 + bk) * NCOLS + colBase + bc;
    v4f b4 = *(const v4f*)bg;
#pragma unroll
    for (int i = 0; i < 4; ++i) Bs[bc + i][bk] = (_Float16)b4[i];

    __syncthreads();

    // ---- fragments from LDS (ISA 7.12.2 layouts) ----
    // A: lane row = wrow+l16; K runs [8*hiK, +8) and [16+8*hiK, +8)
    v8h alo = *(const v8h*)&As[wrow + l16][8 * hiK];
    v8h ahi = *(const v8h*)&As[wrow + l16][16 + 8 * hiK];
    // B: lane col = wcol+l16; K runs [8*hiK, +8) and [16+8*hiK, +8)
    v8h b0 = *(const v8h*)&Bs[wcol + l16][8 * hiK];
    v8h b1 = *(const v8h*)&Bs[wcol + l16][16 + 8 * hiK];

    v16h af, bf;
#pragma unroll
    for (int e = 0; e < 8; ++e) {
      af[e] = alo[e]; af[e + 8] = ahi[e];
      bf[e] = b0[e];  bf[e + 8] = b1[e];
    }

    acc = __builtin_amdgcn_wmma_f32_16x16x32_f16(
        false, af, false, bf, (short)0, acc, false, false);

    __syncthreads();
  }

  const int col = colBase + wcol + l16;
  const float bvv = bias[col];
  // C/D 16x16 f32 layout: row = r + 8*hiK, col = l16
#pragma unroll
  for (int r = 0; r < 8; ++r) {
    const int orow = rowBase + wrow + r + 8 * hiK;
    C[(size_t)orow * NCOLS + col] = acc[r] + bvv;
  }
}

// ---------------------------------------------------------------------------
// Sampling: softmax over L*P logits, 1-D linear interpolation gather from
// value, weighted reduce over (l,p). One wave per (n,q,h); lane = dh channel
// so every gathered row is a contiguous 128-byte coalesced load.
// ---------------------------------------------------------------------------
__global__ __launch_bounds__(256)
void ms_sample(const float* __restrict__ off, const float* __restrict__ logits,
               const float* __restrict__ refp, const float* __restrict__ value,
               const int* __restrict__ shapes, const int* __restrict__ starts,
               float* __restrict__ mid) {
  const int wave = blockIdx.x * (blockDim.x >> 5) + (threadIdx.x >> 5);
  const int lane = threadIdx.x & 31;
  if (wave >= MROWS * HH) return;                // wave-uniform

  const int h  = wave & (HH - 1);
  const int nq = wave / HH;                      // n*LQ + q
  const int n  = nq / LQn;

  const float* lg = logits + (size_t)nq * (HH * LL * PP) + h * (LL * PP);
  const float* of = off    + (size_t)nq * (HH * LL * PP) + h * (LL * PP);

  float w[LL * PP];
  float mx = lg[0];
#pragma unroll
  for (int j = 1; j < LL * PP; ++j) mx = fmaxf(mx, lg[j]);
  float s = 0.f;
#pragma unroll
  for (int j = 0; j < LL * PP; ++j) { w[j] = __expf(lg[j] - mx); s += w[j]; }
  const float inv = 1.f / s;

  float acc = 0.f;
#pragma unroll
  for (int l = 0; l < LL; ++l) {
    const int   Ti = shapes[l];
    const float T  = (float)Ti;
    const int   st = starts[l];
    const float r  = refp[(size_t)nq * LL + l];
#pragma unroll
    for (int p = 0; p < PP; ++p) {
      const int j = l * PP + p;
      const float c  = r * T + of[j] - 0.5f;       // loc*T - 0.5
      const float fl = floorf(c);
      const float wg = c - fl;
      const int i0   = (int)fl;
      const int i0c  = min(max(i0, 0), Ti - 1);
      const int i1c  = min(max(i0 + 1, 0), Ti - 1);
      const float* v0 = value + ((size_t)(n * LIN + st + i0c)) * DD + h * DHH;
      const float* v1 = value + ((size_t)(n * LIN + st + i1c)) * DD + h * DHH;
      acc += (w[j] * inv) * ((1.f - wg) * v0[lane] + wg * v1[lane]);
    }
  }
  mid[(size_t)nq * DD + h * DHH + lane] = acc;
}

// ---------------------------------------------------------------------------
extern "C" void kernel_launch(void* const* d_in, const int* in_sizes, int n_in,
                              void* d_out, int out_size, void* d_ws, size_t ws_size,
                              hipStream_t stream) {
  const float* query  = (const float*)d_in[0];
  const float* refp   = (const float*)d_in[1];
  const float* inflat = (const float*)d_in[2];
  const int*   shapes = (const int*)d_in[3];
  const int*   starts = (const int*)d_in[4];
  const float* W_off  = (const float*)d_in[5];
  const float* b_off  = (const float*)d_in[6];
  const float* W_attn = (const float*)d_in[7];
  const float* b_attn = (const float*)d_in[8];
  const float* W_val  = (const float*)d_in[9];
  const float* b_val  = (const float*)d_in[10];
  const float* W_out  = (const float*)d_in[11];
  const float* b_out  = (const float*)d_in[12];
  float* out = (float*)d_out;

  float* v_val = (float*)d_ws;                         // [15360 x 256]
  float* v_off = v_val + (size_t)MROWS * DD;           // [15360 x 128]
  float* v_att = v_off + (size_t)MROWS * (DD / 2);     // [15360 x 128]
  float* v_mid = v_att + (size_t)MROWS * (DD / 2);     // [15360 x 256]

  const int mBlocks = MROWS / 64;                       // 240

  // 1) value = input_flatten @ W_val + b_val           (15360 x 256 x 256)
  hipLaunchKernelGGL((gemm_wmma_lds<DD>), dim3(mBlocks * (DD / 32)), dim3(256),
                     0, stream, inflat, W_val, b_val, v_val);
  // 2) off = query @ W_off + b_off                     (15360 x 128 x 256)
  hipLaunchKernelGGL((gemm_wmma_lds<DD / 2>), dim3(mBlocks * (DD / 64)), dim3(256),
                     0, stream, query, W_off, b_off, v_off);
  // 3) attn logits = query @ W_attn + b_attn           (15360 x 128 x 256)
  hipLaunchKernelGGL((gemm_wmma_lds<DD / 2>), dim3(mBlocks * (DD / 64)), dim3(256),
                     0, stream, query, W_attn, b_attn, v_att);
  // 4) softmax + gather + weighted reduce -> mid       (one wave per n,q,h)
  {
    int waves = MROWS * HH;
    hipLaunchKernelGGL(ms_sample, dim3((waves + 7) / 8), dim3(256), 0, stream,
                       v_off, v_att, refp, v_val, shapes, starts, v_mid);
  }
  // 5) out = mid @ W_out + b_out                       (15360 x 256 x 256)
  hipLaunchKernelGGL((gemm_wmma_lds<DD>), dim3(mBlocks * (DD / 32)), dim3(256),
                     0, stream, v_mid, W_out, b_out, out);
}
